// MLA_68788196212930
// MI455X (gfx1250) — matrix operand order repaired and checked by reading, not compiled, not run on hardware
//
#include <hip/hip_runtime.h>
#include <cstdint>
#include <cstddef>

// ---------------------------------------------------------------------------
// MLA forward for gfx1250 (MI455X): bf16 WMMA GEMMs (async-LDS double-buffered,
// BK=64) + one-wave flash attention with transposed-V LDS tiles.
// ---------------------------------------------------------------------------

typedef __attribute__((ext_vector_type(16))) __bf16 v16bf;
typedef __attribute__((ext_vector_type(8)))  __bf16 v8bf;
typedef __attribute__((ext_vector_type(8)))  float  v8f;

#define B_   2
#define S_   2048
#define D_   2048
#define DL_  512
#define H_   16
#define DH_  128
#define DN_  64

static __device__ __forceinline__ __bf16 f2bf(float f) {
  unsigned int u = __builtin_bit_cast(unsigned int, f);
  unsigned int r = (u + 0x7FFFu + ((u >> 16) & 1u)) >> 16;  // RNE
  return __builtin_bit_cast(__bf16, (unsigned short)r);
}

// CDNA5 async DMA: global -> LDS, 16 bytes per lane, tracked by ASYNCcnt.
static __device__ __forceinline__ void async_copy16(unsigned int lds_off,
                                                    const void* g) {
  asm volatile("global_load_async_to_lds_b128 %0, %1, off"
               :: "v"(lds_off), "v"((unsigned long long)(uintptr_t)g)
               : "memory");
}
static __device__ __forceinline__ void wait_async0() {
#if defined(__has_builtin) && __has_builtin(__builtin_amdgcn_s_wait_asynccnt)
  __builtin_amdgcn_s_wait_asynccnt(0);
#else
  asm volatile("s_wait_asynccnt 0" ::: "memory");
#endif
}
static __device__ __forceinline__ unsigned int lds_off_of(const void* p) {
  // Generic LDS pointers carry the byte offset in the low 32 bits (ISA 10.2).
  return (unsigned int)(uintptr_t)p;
}
static __device__ __forceinline__ v16bf cat8(v8bf a, v8bf b) {
  return __builtin_shufflevector(a, b, 0, 1, 2, 3, 4, 5, 6, 7,
                                 8, 9, 10, 11, 12, 13, 14, 15);
}

// ---------------------------------------------------------------------------
// fp32 -> bf16 converter
// ---------------------------------------------------------------------------
__global__ void f32_to_bf16_k(const float* __restrict__ in,
                              __bf16* __restrict__ out, int n4) {
  int i = blockIdx.x * blockDim.x + threadIdx.x;
  if (i >= n4) return;
  float4 x = ((const float4*)in)[i];
  __bf16* o = out + (size_t)i * 4;
  o[0] = f2bf(x.x); o[1] = f2bf(x.y); o[2] = f2bf(x.z); o[3] = f2bf(x.w);
}

// V slice of fused KV fp32 buffer -> bf16, transposed to vt[b][h][d][s]
__global__ void transpose_v_k(const float* __restrict__ kv,
                              __bf16* __restrict__ vt, int total) {
  int i = blockIdx.x * blockDim.x + threadIdx.x;  // over BS*HD
  if (i >= total) return;
  int row = i >> 11;            // b*S + s
  int col = i & 2047;           // h*DH + d
  int b = row >> 11;
  int s = row & 2047;
  int h = col >> 7;
  int d = col & 127;
  vt[(((size_t)(b * H_ + h)) * DH_ + d) * S_ + s] =
      f2bf(kv[(size_t)row * 4096 + 2048 + col]);
}

// ---------------------------------------------------------------------------
// RMSNorm over rows of DL=512, emit bf16
// ---------------------------------------------------------------------------
__global__ __launch_bounds__(128)
void rmsnorm_to_bf16_k(const float* __restrict__ in, const float* __restrict__ w,
                       __bf16* __restrict__ out) {
  __shared__ float red[128];
  const int row = blockIdx.x;
  const int t = threadIdx.x;
  float4 x = ((const float4*)(in + (size_t)row * DL_))[t];
  red[t] = x.x * x.x + x.y * x.y + x.z * x.z + x.w * x.w;
  __syncthreads();
  for (int s = 64; s > 0; s >>= 1) {
    if (t < s) red[t] += red[t + s];
    __syncthreads();
  }
  float r = rsqrtf(red[0] / (float)DL_ + 1e-6f);
  float4 wv = ((const float4*)w)[t];
  __bf16* op = out + (size_t)row * DL_ + t * 4;
  op[0] = f2bf(x.x * r * wv.x);
  op[1] = f2bf(x.y * r * wv.y);
  op[2] = f2bf(x.z * r * wv.z);
  op[3] = f2bf(x.w * r * wv.w);
}

// ---------------------------------------------------------------------------
// Head-indexed RoPE (reference quirk: freqs indexed by HEAD) on d in [DN,DH),
// passthrough elsewhere, emit bf16. dst row stride = H*DH.
// ---------------------------------------------------------------------------
__global__ void rope_convert_k(const float* __restrict__ src, int src_stride,
                               const float* __restrict__ freqs,
                               __bf16* __restrict__ dst, int total_pairs) {
  int p = blockIdx.x * blockDim.x + threadIdx.x;
  if (p >= total_pairs) return;
  int row = p >> 10;                 // 1024 pairs per row
  int rem = p & 1023;
  int h   = rem >> 6;
  int dp  = rem & 63;
  int d0  = dp * 2;
  const float* s = src + (size_t)row * src_stride + h * DH_ + d0;
  float x0 = s[0], x1 = s[1], y0, y1;
  if (d0 < DN_) {
    y0 = x0; y1 = x1;
  } else {
    int j = dp - (DN_ / 2);
    float c  = freqs[h * 64 + 2 * j];
    float sn = freqs[h * 64 + 2 * j + 1];
    y0 = x0 * c - x1 * sn;
    y1 = x0 * sn + x1 * c;
  }
  __bf16* d = dst + (size_t)row * (H_ * DH_) + h * DH_ + d0;
  d[0] = f2bf(y0); d[1] = f2bf(y1);
}

// ---------------------------------------------------------------------------
// bf16 WMMA GEMM: C[M x N] = A[M x K] * W[N x K]^T (+ bias), fp32 out.
// Block tile 128(M) x 128(N) x 64(K); 8 waves in 4x2; wave tile 32x64.
// Async-to-LDS double buffering; fragment preload before 8-WMMA bursts.
// ---------------------------------------------------------------------------
__global__ __launch_bounds__(256)
void gemm_bf16_wmma_k(const __bf16* __restrict__ A, const __bf16* __restrict__ W,
                      float* __restrict__ C, const float* __restrict__ bias,
                      int M, int N, int K) {
  __shared__ __align__(16) __bf16 lds_a[2][128 * 64];
  __shared__ __align__(16) __bf16 lds_b[2][128 * 64];
  const int tid  = threadIdx.x;
  const int lane = tid & 31;
  const int wave = tid >> 5;
  const int m_blk = blockIdx.y * 128;
  const int n_blk = blockIdx.x * 128;
  const int m0w = (wave & 3) * 32;     // 4 row-tiles of 32
  const int n0w = (wave >> 2) * 64;    // 2 col-tiles of 64
  const int hi  = (lane & 16) ? 1 : 0;
  const int l15 = lane & 15;
  const int lo  = hi * 8;

  v8f acc[2][4];
#pragma unroll
  for (int am = 0; am < 2; ++am)
#pragma unroll
    for (int t = 0; t < 4; ++t)
#pragma unroll
      for (int i = 0; i < 8; ++i) acc[am][t][i] = 0.0f;

  // staging: 128x64 bf16 per tile = 8192 elems = 1024 16B chunks; 4 per thread.
  // chunk = i*256 + tid -> row = chunk/8, col = (chunk%8)*8 (coalesced: 8
  // consecutive threads cover one 128B row).
  int srow[4], scol[4];
#pragma unroll
  for (int i = 0; i < 4; ++i) {
    int chunk = i * 256 + tid;
    srow[i] = chunk >> 3;
    scol[i] = (chunk & 7) * 8;
  }
  const unsigned int la_base[2] = {lds_off_of(&lds_a[0][0]),
                                   lds_off_of(&lds_a[1][0])};
  const unsigned int lb_base[2] = {lds_off_of(&lds_b[0][0]),
                                   lds_off_of(&lds_b[1][0])};

  auto issue = [&](int buf, int k0) {
#pragma unroll
    for (int i = 0; i < 4; ++i) {
      unsigned int loff = (unsigned)(srow[i] * 64 + scol[i]) * 2;
      async_copy16(la_base[buf] + loff,
                   &A[(size_t)(m_blk + srow[i]) * K + k0 + scol[i]]);
      async_copy16(lb_base[buf] + loff,
                   &W[(size_t)(n_blk + srow[i]) * K + k0 + scol[i]]);
    }
  };

  issue(0, 0);
  int cur = 0;
  for (int k0 = 0; k0 < K; k0 += 64) {
    wait_async0();
    __syncthreads();
    if (k0 + 64 < K) issue(cur ^ 1, k0 + 64);

    const __bf16* la = &lds_a[cur][0];
    const __bf16* lb = &lds_b[cur][0];
#pragma unroll
    for (int ks = 0; ks < 64; ks += 32) {
      // preload all fragments, then an uninterrupted 8-WMMA burst
      v16bf af[2], bf[4];
#pragma unroll
      for (int am = 0; am < 2; ++am) {
        const __bf16* ap = la + (m0w + am * 16 + l15) * 64 + ks;
        af[am] = cat8(*(const v8bf*)(ap + lo), *(const v8bf*)(ap + 16 + lo));
      }
#pragma unroll
      for (int nt = 0; nt < 4; ++nt) {
        const __bf16* bp = lb + (n0w + nt * 16 + l15) * 64 + ks + lo;
        bf[nt] = cat8(*(const v8bf*)(bp), *(const v8bf*)(bp + 16));
      }
#pragma unroll
      for (int nt = 0; nt < 4; ++nt)
#pragma unroll
        for (int am = 0; am < 2; ++am)
          acc[am][nt] = __builtin_amdgcn_wmma_f32_16x16x32_bf16(
              false, af[am], false, bf[nt], (short)0, acc[am][nt], false, false);
    }
    __syncthreads();
    cur ^= 1;
  }

#pragma unroll
  for (int am = 0; am < 2; ++am)
#pragma unroll
    for (int nt = 0; nt < 4; ++nt)
#pragma unroll
      for (int r = 0; r < 8; ++r) {
        int row = m_blk + m0w + am * 16 + (hi ? r + 8 : r);
        int col = n_blk + n0w + nt * 16 + l15;
        float v = acc[am][nt][r];
        if (bias) v += bias[col];
        C[(size_t)row * N + col] = v;
      }
}

// ---------------------------------------------------------------------------
// Flash attention: one wave (32 threads) per (b, h, 16-query-row tile).
// V is pre-transposed in global: vt[b][h][d][s] so the LDS tile is d-major and
// each P x V B-fragment is two contiguous ds_load_b128.
// ---------------------------------------------------------------------------
__global__ __launch_bounds__(32)
void mla_attention_k(const __bf16* __restrict__ q, const __bf16* __restrict__ k,
                     const __bf16* __restrict__ vt, __bf16* __restrict__ o) {
  __shared__ __align__(16) __bf16 lds_vt[128 * 32];  // [d][kk]
  __shared__ __align__(16) __bf16 lds_p[16 * 32];
  const int lane = threadIdx.x;
  const int qtiles = S_ / 16;
  int id = blockIdx.x;
  const int b = id / (H_ * qtiles); id -= b * H_ * qtiles;
  const int h = id / qtiles;
  const int q0 = (id % qtiles) * 16;
  const int l15 = lane & 15;
  const int hi = (lane & 16) ? 1 : 0;
  const int lo = hi * 8;
  const float scale = 0.08838834764831845f;  // 1/sqrt(128)

  // Q fragments (A-layout), 4 K-steps of 32 along DH=128
  v16bf qf[4];
  {
    const __bf16* qrow = q + (((size_t)(b * S_ + q0 + l15)) * H_ + h) * DH_;
#pragma unroll
    for (int ks = 0; ks < 4; ++ks)
      qf[ks] = cat8(*(const v8bf*)(qrow + ks * 32 + lo),
                    *(const v8bf*)(qrow + ks * 32 + 16 + lo));
  }

  v8f oacc[8];
#pragma unroll
  for (int t = 0; t < 8; ++t)
#pragma unroll
    for (int i = 0; i < 8; ++i) oacc[t][i] = 0.0f;
  float m_run[8], l_run[8];
#pragma unroll
  for (int r = 0; r < 8; ++r) { m_run[r] = -1e30f; l_run[r] = 0.0f; }

  const unsigned int lvbase = lds_off_of(&lds_vt[0]);
  const __bf16* vtbh = vt + ((size_t)(b * H_ + h)) * DH_ * S_;
  const int nblk = (q0 + 15) / 32 + 1;  // causal: kv blocks of 32
  for (int blk = 0; blk < nblk; ++blk) {
    const int kv0 = blk * 32;
    // async-stage V^T tile (128 d-rows x 32 kv) into LDS; 16 chunks per lane,
    // 4 consecutive lanes cover one 64B d-row (coalesced).
    {
#pragma unroll
      for (int i = 0; i < 16; ++i) {
        int chunk = i * 32 + lane;
        int d = chunk >> 2;
        int part = (chunk & 3) * 8;
        async_copy16(lvbase + (unsigned)(d * 32 + part) * 2,
                     vtbh + (size_t)d * S_ + kv0 + part);
      }
      wait_async0();
    }
    __syncthreads();

    // scores S = Q x K^T for two 16-wide column tiles (B frags from global)
    float sc[2][8];
#pragma unroll
    for (int ct = 0; ct < 2; ++ct) {
      v8f a;
#pragma unroll
      for (int i = 0; i < 8; ++i) a[i] = 0.0f;
      const int krow = kv0 + ct * 16 + l15;
      const __bf16* kr = k + (((size_t)(b * S_ + krow)) * H_ + h) * DH_;
#pragma unroll
      for (int ks = 0; ks < 4; ++ks) {
        const __bf16* kp = kr + ks * 32 + lo;
        v16bf bfrag = cat8(*(const v8bf*)(kp), *(const v8bf*)(kp + 16));
        a = __builtin_amdgcn_wmma_f32_16x16x32_bf16(
            false, qf[ks], false, bfrag, (short)0, a, false, false);
      }
#pragma unroll
      for (int r = 0; r < 8; ++r) sc[ct][r] = a[r];
    }

    // scale + causal mask + online softmax (row stats per lane-half)
#pragma unroll
    for (int r = 0; r < 8; ++r) {
      const int qg = q0 + (hi ? r + 8 : r);
#pragma unroll
      for (int ct = 0; ct < 2; ++ct) {
        const int kg = kv0 + ct * 16 + l15;
        float x = sc[ct][r] * scale;
        sc[ct][r] = (kg <= qg) ? x : -1e30f;
      }
    }
#pragma unroll
    for (int r = 0; r < 8; ++r) {
      float mx = fmaxf(sc[0][r], sc[1][r]);
#pragma unroll
      for (int msk = 1; msk < 16; msk <<= 1)
        mx = fmaxf(mx, __shfl_xor(mx, msk, 32));
      float mnew  = fmaxf(m_run[r], mx);
      float alpha = __expf(m_run[r] - mnew);
      float p0 = __expf(sc[0][r] - mnew);
      float p1 = __expf(sc[1][r] - mnew);
      sc[0][r] = p0; sc[1][r] = p1;
      float s = p0 + p1;
#pragma unroll
      for (int msk = 1; msk < 16; msk <<= 1)
        s += __shfl_xor(s, msk, 32);
      l_run[r] = l_run[r] * alpha + s;
      m_run[r] = mnew;
#pragma unroll
      for (int t = 0; t < 8; ++t) oacc[t][r] *= alpha;
    }

    // P (bf16) -> LDS row-major, reload as A-fragment
#pragma unroll
    for (int r = 0; r < 8; ++r) {
      const int mrow = hi ? r + 8 : r;
      lds_p[mrow * 32 + l15]      = f2bf(sc[0][r]);
      lds_p[mrow * 32 + 16 + l15] = f2bf(sc[1][r]);
    }
    __syncthreads();

    v16bf pf;
    {
      const __bf16* pp = &lds_p[l15 * 32];
      pf = cat8(*(const v8bf*)(pp + lo), *(const v8bf*)(pp + 16 + lo));
    }

    // O += P x V over 8 d-tiles of 16 (B frag = 2 contiguous ds_load_b128)
#pragma unroll
    for (int dt = 0; dt < 8; ++dt) {
      const __bf16* vp = &lds_vt[(dt * 16 + l15) * 32 + lo];
      v16bf vf = cat8(*(const v8bf*)(vp), *(const v8bf*)(vp + 16));
      oacc[dt] = __builtin_amdgcn_wmma_f32_16x16x32_bf16(
          false, pf, false, vf, (short)0, oacc[dt], false, false);
    }
    __syncthreads();
  }

  // normalize and store O (bf16)
#pragma unroll
  for (int r = 0; r < 8; ++r) {
    const int mrow = hi ? r + 8 : r;
    const float inv = 1.0f / l_run[r];
    __bf16* orow = o + (((size_t)(b * S_ + q0 + mrow)) * H_ + h) * DH_;
#pragma unroll
    for (int dt = 0; dt < 8; ++dt)
      orow[dt * 16 + l15] = f2bf(oacc[dt][r] * inv);
  }
}

// ---------------------------------------------------------------------------
// Host orchestration
// ---------------------------------------------------------------------------
extern "C" void kernel_launch(void* const* d_in, const int* in_sizes, int n_in,
                              void* d_out, int out_size, void* d_ws, size_t ws_size,
                              hipStream_t stream) {
  (void)in_sizes; (void)n_in; (void)out_size; (void)ws_size;
  const float* x      = (const float*)d_in[0];
  const float* freqs  = (const float*)d_in[1];
  const float* wq_w   = (const float*)d_in[2];
  const float* wq_b   = (const float*)d_in[3];
  const float* wkva_w = (const float*)d_in[4];
  const float* wkvb_w = (const float*)d_in[5];
  const float* norm_w = (const float*)d_in[6];
  const float* out_w  = (const float*)d_in[7];
  const float* out_b  = (const float*)d_in[8];
  float* out = (float*)d_out;

  const int BS = B_ * S_;       // 4096
  const int HD = H_ * DH_;      // 2048

  char* ws = (char*)d_ws;
  size_t off = 0;
  auto alloc = [&](size_t bytes) -> char* {
    char* p = ws + off;
    off += (bytes + 255) & ~(size_t)255;
    return p;
  };
  __bf16* x_bf    = (__bf16*)alloc((size_t)BS * D_ * 2);
  __bf16* wq_bf   = (__bf16*)alloc((size_t)HD * D_ * 2);
  __bf16* wkva_bf = (__bf16*)alloc((size_t)DL_ * D_ * 2);
  __bf16* wkvb_bf = (__bf16*)alloc((size_t)2 * HD * DL_ * 2);
  __bf16* outw_bf = (__bf16*)alloc((size_t)D_ * HD * 2);
  __bf16* lat_bf  = (__bf16*)alloc((size_t)BS * DL_ * 2);
  __bf16* q_bf    = (__bf16*)alloc((size_t)BS * HD * 2);
  __bf16* k_bf    = (__bf16*)alloc((size_t)BS * HD * 2);
  __bf16* vt_bf   = (__bf16*)alloc((size_t)BS * HD * 2);   // transposed V
  __bf16* o_bf    = (__bf16*)alloc((size_t)BS * HD * 2);
  float*  tmp     = (float*)alloc((size_t)BS * 2 * HD * 4);  // reused fp32 scratch

  auto cvt = [&](const float* src, __bf16* dst, size_t n) {
    int n4 = (int)(n / 4);
    f32_to_bf16_k<<<(n4 + 255) / 256, 256, 0, stream>>>(src, dst, n4);
  };
  cvt(x, x_bf, (size_t)BS * D_);
  cvt(wq_w, wq_bf, (size_t)HD * D_);
  cvt(wkva_w, wkva_bf, (size_t)DL_ * D_);
  cvt(wkvb_w, wkvb_bf, (size_t)2 * HD * DL_);
  cvt(out_w, outw_bf, (size_t)D_ * HD);

  // lat_raw = x @ wkv_a^T   (M=4096, N=512, K=2048)
  gemm_bf16_wmma_k<<<dim3(DL_ / 128, BS / 128), 256, 0, stream>>>(
      x_bf, wkva_bf, tmp, nullptr, BS, DL_, D_);
  rmsnorm_to_bf16_k<<<BS, 128, 0, stream>>>(tmp, norm_w, lat_bf);

  // kv = lat @ wkv_b^T      (M=4096, N=4096, K=512)
  gemm_bf16_wmma_k<<<dim3((2 * HD) / 128, BS / 128), 256, 0, stream>>>(
      lat_bf, wkvb_bf, tmp, nullptr, BS, 2 * HD, DL_);
  {
    int tp = BS * (HD / 2);
    rope_convert_k<<<(tp + 255) / 256, 256, 0, stream>>>(
        tmp, 2 * HD, freqs, k_bf, tp);
    int tv = BS * HD;
    transpose_v_k<<<(tv + 255) / 256, 256, 0, stream>>>(tmp, vt_bf, tv);
  }

  // q = x @ wq^T + wq_b     (M=4096, N=2048, K=2048)
  gemm_bf16_wmma_k<<<dim3(HD / 128, BS / 128), 256, 0, stream>>>(
      x_bf, wq_bf, tmp, wq_b, BS, HD, D_);
  {
    int tp = BS * (HD / 2);
    rope_convert_k<<<(tp + 255) / 256, 256, 0, stream>>>(
        tmp, HD, freqs, q_bf, tp);
  }

  // flash attention: one wave per (b, h, 16-row q tile)
  mla_attention_k<<<B_ * H_ * (S_ / 16), 32, 0, stream>>>(q_bf, k_bf, vt_bf, o_bf);

  // out = o @ out_w^T + out_b  (M=4096, N=2048, K=2048) -> fp32 d_out
  gemm_bf16_wmma_k<<<dim3(D_ / 128, BS / 128), 256, 0, stream>>>(
      o_bf, outw_bf, out, out_b, BS, D_, HD);
}
